// SSIMLoss_73787538145643
// MI455X (gfx1250) — matrix operand order repaired and checked by reading, not compiled, not run on hardware
//
#include <hip/hip_runtime.h>

// SSIM loss (16x3x512x512 f32 pairs) fused into one pass + tiny reduce.
// Roofline: ~101 MB HBM traffic -> ~4.3us floor @ 23.3 TB/s. Scalar-f32 conv
// would be ~3 GFLOP and VALU-bound, so the 5 separable 11-tap blurs are routed
// through v_wmma_f32_16x16x32_f16 with banded Gaussian operand matrices
// (K=32 covers the full 26-wide receptive field in one instruction).
// Each wave produces a 16x32 output tile: 3 vertical + 2 horizontal WMMAs per
// field, 25 WMMAs per wave, squared/cross fields derived with v_pk_mul_f16.

typedef __attribute__((ext_vector_type(16))) _Float16 v16h;
typedef __attribute__((ext_vector_type(8)))  _Float16 v8h;
typedef __attribute__((ext_vector_type(8)))  float    v8f;

#define IMG   512
#define OUTW  502          // 512 - 11 + 1
#define KS    11
#define CST   40           // staging col stride (halves): 80B, 16B-aligned
#define VST   56           // vert-tile row stride (halves): 112B, 16B-aligned, conflict-free
#define WPB   4            // waves per block (128 threads)
#define NBLK  6144         // 48 planes * 32 row-tiles * 16 col-tiles / 4 waves
#define C1F   1.0e-4f
#define C2F   9.0e-4f

// Normalized 11-tap Gaussian, sigma=1.5 (precomputed: exp(-(i-5)^2/4.5)/sum).
__device__ __constant__ float G11[KS] = {
  0.00102839f, 0.00759876f, 0.03600077f, 0.10936070f, 0.21300549f,
  0.26601230f,
  0.21300549f, 0.10936070f, 0.03600077f, 0.00759876f, 0.00102839f
};

static __device__ __forceinline__ v8f zero8() {
  v8f z;
  #pragma unroll
  for (int i = 0; i < 8; ++i) z[i] = 0.0f;
  return z;
}

// 16 halves as two aligned 16B pieces (ds_load_b128 x2 after inlining).
static __device__ __forceinline__ v16h ld16(const _Float16* p0, const _Float16* p1) {
  v8h a = *(const v8h*)p0;
  v8h b = *(const v8h*)p1;
  v16h r;
  #pragma unroll
  for (int i = 0; i < 8; ++i) { r[i] = a[i]; r[i + 8] = b[i]; }
  return r;
}

__global__ __launch_bounds__(128) void ssim_wmma_kernel(const float* __restrict__ X,
                                                        const float* __restrict__ Y,
                                                        float* __restrict__ partial) {
  __shared__ __align__(16) _Float16 sx[WPB][48 * CST];  // x patch, column-major f16
  __shared__ __align__(16) _Float16 sy[WPB][48 * CST];  // y patch, column-major f16
  __shared__ __align__(16) _Float16 sv[WPB][16 * VST];  // vertical-pass bounce tile
  __shared__ float sred[128];

  const int tid  = threadIdx.x;
  const int w    = tid >> 5;
  const int lane = tid & 31;
  const int hi   = lane >> 4;
  const int ln   = lane & 15;

  // A operand: banded vertical Gaussian G[r,k] = g[k-r] (16x32 f16).
  // A-layout (16-bit 16x32): lanes L/L+16 hold row M=L&15;
  //   k = elem + 16*(lane>=16) (contiguous, matches ldB)
  //
  // B operand: banded horizontal Gaussian H[k,n] = g[k-n] (32x16 f16),
  // table layout: n = lane&15, k = elem + 8*(elem>=8) + 8*(lane>=16).
  v16h Ag, Bh;
  #pragma unroll
  for (int e = 0; e < 16; ++e) {
    int ka = e + hi * 16;
    int da = ka - ln;
    int ia = da < 0 ? 0 : (da > 10 ? 10 : da);
    Ag[e] = (_Float16)((da >= 0 && da < KS) ? G11[ia] : 0.0f);
    int kb = e + (e >= 8 ? 8 : 0) + hi * 8;
    int db = kb - ln;
    int ib = db < 0 ? 0 : (db > 10 ? 10 : db);
    Bh[e] = (_Float16)((db >= 0 && db < KS) ? G11[ib] : 0.0f);
  }

  // Tile decode: 48 planes x (32 row-tiles of 16) x (16 col-tiles of 32).
  const int tile  = blockIdx.x * WPB + w;
  const int plane = tile >> 9;
  const int rem   = tile & 511;
  const int r0    = (rem >> 4) << 4;
  const int c0    = (rem & 15) << 5;
  const float* xp = X + (size_t)plane * (IMG * IMG);
  const float* yp = Y + (size_t)plane * (IMG * IMG);

  _Float16* lx = sx[w];
  _Float16* ly = sy[w];
  _Float16* lv = sv[w];

  // Stage 26x42 patch (zero-padded to 32x48) column-major f16; per-wave LDS
  // region, DS ops are in-order within a wave -> no barrier needed.
  for (int rr = 0; rr < 32; ++rr) {
    const int  gr  = r0 + rr;
    const bool rok = (rr < 26) && (gr < IMG);
    {
      int cc = lane;                      // 0..31 (< 42 always)
      int gc = c0 + cc;
      bool ok = rok && (gc < IMG);
      float xv = 0.0f, yv = 0.0f;
      if (ok) { size_t gi = (size_t)gr * IMG + gc; xv = xp[gi]; yv = yp[gi]; }
      lx[cc * CST + rr] = (_Float16)xv;
      ly[cc * CST + rr] = (_Float16)yv;
    }
    if (lane < 16) {
      int cc = 32 + lane;                 // 32..47; >=42 forced to zero
      int gc = c0 + cc;
      bool ok = rok && (cc < 42) && (gc < IMG);
      float xv = 0.0f, yv = 0.0f;
      if (ok) { size_t gi = (size_t)gr * IMG + gc; xv = xp[gi]; yv = yp[gi]; }
      lx[cc * CST + rr] = (_Float16)xv;
      ly[cc * CST + rr] = (_Float16)yv;
    }
  }

  // B operands: col = cg*16 + ln, rows k0 = hi*16 .. +15 (contiguous 32B).
  auto ldB = [&](const _Float16* base, int cg) -> v16h {
    const _Float16* p = base + (cg * 16 + ln) * CST + hi * 16;
    return ld16(p, p + 8);
  };
  const v16h bx0 = ldB(lx, 0), bx1 = ldB(lx, 1), bx2 = ldB(lx, 2);
  const v16h by0 = ldB(ly, 0), by1 = ldB(ly, 1), by2 = ldB(ly, 2);

  struct P2 { v8f a, b; };
  // One field: 3 vertical WMMAs -> f16 bounce (16x48) -> 2 horizontal WMMAs.
  auto blur = [&](v16h b0, v16h b1, v16h b2) -> P2 {
    v8f z  = zero8();
    v8f d0 = __builtin_amdgcn_wmma_f32_16x16x32_f16(false, Ag, false, b0, (short)0, z, false, false);
    v8f d1 = __builtin_amdgcn_wmma_f32_16x16x32_f16(false, Ag, false, b1, (short)0, z, false, false);
    v8f d2 = __builtin_amdgcn_wmma_f32_16x16x32_f16(false, Ag, false, b2, (short)0, z, false, false);
    // C/D layout: elem v -> row M = v + hi*8, col N = ln (+16/+32 per group).
    #pragma unroll
    for (int v = 0; v < 8; ++v) {
      int row = v + hi * 8;
      lv[row * VST + ln]      = (_Float16)d0[v];
      lv[row * VST + ln + 16] = (_Float16)d1[v];
      lv[row * VST + ln + 32] = (_Float16)d2[v];
    }
    P2 r;
    {
      const _Float16* p = &lv[ln * VST + hi * 8];        // og=0: cols K(e)
      v16h a2 = ld16(p, p + 16);
      r.a = __builtin_amdgcn_wmma_f32_16x16x32_f16(false, a2, false, Bh, (short)0, z, false, false);
    }
    {
      const _Float16* p = &lv[ln * VST + 16 + hi * 8];   // og=1: cols 16+K(e)
      v16h a2 = ld16(p, p + 16);
      r.b = __builtin_amdgcn_wmma_f32_16x16x32_f16(false, a2, false, Bh, (short)0, z, false, false);
    }
    return r;
  };

  // Five blurred moments; squared/cross operands derived via v_pk_mul_f16.
  P2 mux = blur(bx0, bx1, bx2);
  P2 muy = blur(by0, by1, by2);
  P2 exx = blur(bx0 * bx0, bx1 * bx1, bx2 * bx2);
  P2 eyy = blur(by0 * by0, by1 * by1, by2 * by2);
  P2 exy = blur(bx0 * by0, bx1 * by1, bx2 * by2);

  float lsum = 0.0f;
  #pragma unroll
  for (int og = 0; og < 2; ++og) {
    v8f mx8 = (og == 0) ? mux.a : mux.b;
    v8f my8 = (og == 0) ? muy.a : muy.b;
    v8f xx8 = (og == 0) ? exx.a : exx.b;
    v8f yy8 = (og == 0) ? eyy.a : eyy.b;
    v8f xy8 = (og == 0) ? exy.a : exy.b;
    int ocol = c0 + og * 16 + ln;
    #pragma unroll
    for (int v = 0; v < 8; ++v) {
      float mx  = mx8[v], my = my8[v];
      float mx2 = mx * mx, my2 = my * my, mxy = mx * my;
      float vx  = xx8[v] - mx2;
      float vy  = yy8[v] - my2;
      float cv  = xy8[v] - mxy;
      float num = (2.0f * mxy + C1F) * (2.0f * cv + C2F);
      float den = (mx2 + my2 + C1F) * (vx + vy + C2F);
      int orow  = r0 + v + hi * 8;
      if (orow < OUTW && ocol < OUTW) lsum += num / den;
    }
  }

  // Deterministic in-block reduction -> one partial per block (no atomics).
  sred[tid] = lsum;
  __syncthreads();
  for (int off = 64; off > 0; off >>= 1) {
    if (tid < off) sred[tid] += sred[tid + off];
    __syncthreads();
  }
  if (tid == 0) partial[blockIdx.x] = sred[0];
}

__global__ __launch_bounds__(256) void ssim_reduce_kernel(const float* __restrict__ partial,
                                                          float* __restrict__ out) {
  __shared__ float sred[256];
  float s = 0.0f;
  for (int i = threadIdx.x; i < NBLK; i += 256) s += partial[i];
  sred[threadIdx.x] = s;
  __syncthreads();
  for (int off = 128; off > 0; off >>= 1) {
    if (threadIdx.x < off) sred[threadIdx.x] += sred[threadIdx.x + off];
    __syncthreads();
  }
  if (threadIdx.x == 0)
    out[0] = 1.0f - sred[0] / (48.0f * 502.0f * 502.0f);  // mean over 12,096,192 px
}

extern "C" void kernel_launch(void* const* d_in, const int* in_sizes, int n_in,
                              void* d_out, int out_size, void* d_ws, size_t ws_size,
                              hipStream_t stream) {
  const float* X = (const float*)d_in[0];   // input  [16,3,512,512] f32
  const float* Y = (const float*)d_in[1];   // target [16,3,512,512] f32
  float* partial = (float*)d_ws;            // NBLK floats of scratch (fully overwritten)
  ssim_wmma_kernel<<<NBLK, 128, 0, stream>>>(X, Y, partial);
  ssim_reduce_kernel<<<1, 256, 0, stream>>>(partial, (float*)d_out);
}
